// LSTMModelBinary_46239617909098
// MI455X (gfx1250) — matrix operand order repaired and checked by reading, not compiled, not run on hardware
//
#include <hip/hip_runtime.h>
#include <hip/hip_bf16.h>

// ---------------------------------------------------------------------------
// 2-layer LSTM (S=512, B=64, H=256) + sigmoid head for MI455X (gfx1250).
// Serial recurrence (1024 steps) -> persistent single-WGP kernel per layer,
// bf16 WMMA (v_wmma_f32_16x16x32_bf16) for x@W_ih^T and h@W_hh^T,
// h/gates/bias staged in LDS, cell state in registers.
// Weight loads stay inside the timestep loop via an opaque 0 offset (keeps
// global address space -> global_load_b128, not flat), and the inner GEMM is
// software-pipelined (B double-buffered, next-A prefetched).
// ---------------------------------------------------------------------------

typedef __attribute__((ext_vector_type(16))) __bf16 v16bf;
typedef __attribute__((ext_vector_type(8)))  float  v8f;

#define S_LEN 512
#define BATCH 64
#define HID   256
#define NGATE 1024                 // 4*HID
#define FRAG_U32_PER_MAT 131072u   // 64 ntiles * 8 ktiles * 32 lanes * 8 dwords

union Frag {
  uint4 u4[2];
  v16bf v;
};

__device__ __forceinline__ unsigned short f2bf(float f) {
  unsigned int u = __float_as_uint(f);
  u += 0x7FFFu + ((u >> 16) & 1u);          // round-to-nearest-even
  return (unsigned short)(u >> 16);
}
__device__ __forceinline__ float bf2f(unsigned short s) {
  return __uint_as_float(((unsigned int)s) << 16);
}
__device__ __forceinline__ float sigm(float x) { return 1.0f / (1.0f + __expf(-x)); }

// B fragment: 32 contiguous bytes per lane -> two global_load_b128
__device__ __forceinline__ Frag load_fragB(const unsigned int* __restrict__ p) {
  Frag f;
  const uint4* q = reinterpret_cast<const uint4*>(p);
  f.u4[0] = q[0];
  f.u4[1] = q[1];
  return f;
}
// A fragment: two 16B runs 32B apart (16-bit A layout, K runs of 8)
__device__ __forceinline__ Frag load_fragA_g(const unsigned short* __restrict__ p) {
  Frag f;
  const uint4* q = reinterpret_cast<const uint4*>(p);
  f.u4[0] = q[0];
  f.u4[1] = q[2];
  return f;
}
__device__ __forceinline__ Frag load_fragA_s(const unsigned short* p) {
  Frag f;
  const uint4* q = reinterpret_cast<const uint4*>(p);
  f.u4[0] = q[0];
  f.u4[1] = q[2];
  return f;
}

// ---------------------------------------------------------------------------
// Pack W_ih / W_hh (fp32 [L,1024,256]) into bf16 WMMA B-fragment order:
// fragment(lane, v) holds B[k,n]=W[n,k] with n = nt*16 + lane%16,
// k = kt*32 + (v/4)*16 + (lane/16)*8 + 2(v%4) (+e in low/high half of the dword).
// ---------------------------------------------------------------------------
__global__ __launch_bounds__(256) void pack_weights(const float* __restrict__ W_ih,
                                                    const float* __restrict__ W_hh,
                                                    unsigned int* __restrict__ wf) {
  unsigned int tid   = blockIdx.x * 256u + threadIdx.x;  // < 524288
  unsigned int v     = tid & 7u;
  unsigned int lane  = (tid >> 3) & 31u;
  unsigned int kt    = (tid >> 8) & 7u;
  unsigned int nt    = (tid >> 11) & 63u;
  unsigned int mat   = (tid >> 17) & 1u;
  unsigned int layer = (tid >> 18) & 1u;
  unsigned int n = nt * 16u + (lane & 15u);
  unsigned int k = kt * 32u + (v >> 2) * 16u + (lane >> 4) * 8u + (v & 3u) * 2u;
  const float* W = mat ? W_hh : W_ih;
  const float* p = W + layer * (NGATE * HID) + n * HID + k;
  wf[tid] = (unsigned int)f2bf(p[0]) | ((unsigned int)f2bf(p[1]) << 16);
}

// x [B=64, 1, C=512, T=256] fp32 -> time-major bf16 xb[s][b][k]
__global__ __launch_bounds__(256) void convert_x(const float* __restrict__ x,
                                                 unsigned short* __restrict__ xb) {
  unsigned int tid = blockIdx.x * 256u + threadIdx.x;    // < 8388608
  unsigned int k = tid & 255u;
  unsigned int b = (tid >> 8) & 63u;
  unsigned int s = tid >> 14;
  xb[tid] = f2bf(x[b * (S_LEN * HID) + s * HID + k]);
}

__global__ __launch_bounds__(256) void pack_bias(const float* __restrict__ b_ih,
                                                 const float* __restrict__ b_hh,
                                                 float* __restrict__ bias) {
  unsigned int tid = blockIdx.x * 256u + threadIdx.x;    // < 2048
  bias[tid] = b_ih[tid] + b_hh[tid];
}

// ---------------------------------------------------------------------------
// Persistent LSTM layer: 1 workgroup, 1024 threads = 32 waves.
// Wave w: mtile = w&3 (batch rows), ntiles (w>>2)*8 .. +7  -> 8 C tiles/wave.
// Per step: 128 WMMAs/wave; gates -> LDS (bf16); elementwise update with
// c in registers; h -> LDS (bf16) + global sequence output.
// Dynamic LDS: 32KB h + 128KB gates + 4KB bias = 164KB (< 320KB WGP limit).
// ---------------------------------------------------------------------------
__global__ __launch_bounds__(1024) void lstm_layer_k(const unsigned short* __restrict__ xin,
                                                     const unsigned int* __restrict__ wl,
                                                     const float* __restrict__ bias,
                                                     unsigned short* __restrict__ hout) {
  extern __shared__ __align__(16) char smem[];
  unsigned short* sm_h = (unsigned short*)smem;             // [64][256]   32KB
  unsigned short* sm_g = (unsigned short*)(smem + 32768);   // [64][1024] 128KB
  float*          sm_b = (float*)(smem + 163840);           // [1024]       4KB

  const unsigned int tid   = threadIdx.x;
  const unsigned int lane  = tid & 31u;
  const unsigned int w     = tid >> 5;
  const unsigned int mtile = w & 3u;
  const unsigned int ngrp  = w >> 2;
  const unsigned int mrow  = mtile * 16u + (lane & 15u);   // 16-bit A: M = lane%16
  const unsigned int khalf = lane >> 4;                    // K half select
  const unsigned int bofs  = (ngrp * 64u + lane) * 8u;     // lane's base u32 in tile (nt=ngrp/4, kt=2*(ngrp%4))

  // h0 = 0; bias -> LDS
  unsigned int* h32 = (unsigned int*)sm_h;
#pragma unroll
  for (int i = 0; i < 8; ++i) h32[tid + i * 1024] = 0u;
  sm_b[tid] = bias[tid];

  float creg[16];
#pragma unroll
  for (int i = 0; i < 16; ++i) creg[i] = 0.0f;

  __syncthreads();

  unsigned int wofs = 0u;   // always 0, but opaque per-iteration (blocks LICM,
                            // preserves wl's global address space)

#pragma clang loop unroll(disable)
  for (int t = 0; t < S_LEN; ++t) {
    asm volatile("" : "+s"(wofs));
    const unsigned int* wi = wl + wofs;                 // W_ih fragments (global AS)
    const unsigned int* wh = wi + FRAG_U32_PER_MAT;     // W_hh fragments

    const unsigned short* xt = xin + (unsigned int)t * (BATCH * HID);
    const unsigned short* xa = xt + mrow * HID + khalf * 8u;   // lane's A base
    const unsigned short* ha = sm_h + mrow * HID + khalf * 8u;

    if (t + 1 < S_LEN)   // next timestep's A rows (global_prefetch_b8)
      __builtin_prefetch((const void*)(xa + BATCH * HID), 0, 0);

    v8f acc[8];
#pragma unroll
    for (int j = 0; j < 8; ++j) {
      v8f z = {0.f, 0.f, 0.f, 0.f, 0.f, 0.f, 0.f, 0.f};
      acc[j] = z;
    }

    // ---- gates += x_t @ W_ih^T  (pipelined: B double-buffered, next-A early) ----
    {
      Frag a  = load_fragA_g(xa);                 // kt = 0
      Frag bc = load_fragB(wi + bofs);            // (nt0, kt0)
#pragma unroll
      for (int kt = 0; kt < 8; ++kt) {
        Frag an = (kt < 7) ? load_fragA_g(xa + (kt + 1) * 32) : a;
#pragma unroll
        for (int j = 0; j < 8; ++j) {
          Frag bn;
          if (j < 7)       bn = load_fragB(wi + bofs + (((j + 1) * 8u + kt) << 8));
          else if (kt < 7) bn = load_fragB(wi + bofs + ((unsigned)(kt + 1) << 8));
          else             bn = bc;
          acc[j] = __builtin_amdgcn_wmma_f32_16x16x32_bf16(false, a.v, false, bc.v,
                                                           (short)0, acc[j], false, false);
          bc = bn;
        }
        a = an;
      }
    }

    // ---- gates += h_{t-1} @ W_hh^T (A from LDS) ----
    {
      Frag a  = load_fragA_s(ha);
      Frag bc = load_fragB(wh + bofs);
#pragma unroll
      for (int kt = 0; kt < 8; ++kt) {
        Frag an = (kt < 7) ? load_fragA_s(ha + (kt + 1) * 32) : a;
#pragma unroll
        for (int j = 0; j < 8; ++j) {
          Frag bn;
          if (j < 7)       bn = load_fragB(wh + bofs + (((j + 1) * 8u + kt) << 8));
          else if (kt < 7) bn = load_fragB(wh + bofs + ((unsigned)(kt + 1) << 8));
          else             bn = bc;
          acc[j] = __builtin_amdgcn_wmma_f32_16x16x32_bf16(false, a.v, false, bc.v,
                                                           (short)0, acc[j], false, false);
          bc = bn;
        }
        a = an;
      }
    }

    // C tile layout: VGPR v -> row M = mtile*16 + v + 8*(lane/16), col N = nt*16 + lane%16
#pragma unroll
    for (int j = 0; j < 8; ++j) {
      unsigned int ncol = (ngrp * 8u + j) * 16u + (lane & 15u);
#pragma unroll
      for (int v = 0; v < 8; ++v) {
        unsigned int row = mtile * 16u + (unsigned int)v + (khalf << 3);
        sm_g[row * NGATE + ncol] = f2bf(acc[j][v]);
      }
    }
    __syncthreads();

    // ---- elementwise cell update: i,f,g,o at columns j, 256+j, 512+j, 768+j ----
#pragma unroll
    for (int i = 0; i < 16; ++i) {
      unsigned int p  = ((unsigned int)i << 10) + tid;   // 0..16383
      unsigned int b  = p >> 8;
      unsigned int jx = p & 255u;
      const unsigned short* gr = sm_g + b * NGATE;
      float ig = bf2f(gr[jx])        + sm_b[jx];
      float fg = bf2f(gr[256 + jx])  + sm_b[256 + jx];
      float gg = bf2f(gr[512 + jx])  + sm_b[512 + jx];
      float og = bf2f(gr[768 + jx])  + sm_b[768 + jx];
      ig = sigm(ig);
      fg = sigm(fg);
      og = sigm(og);
      gg = tanhf(gg);
      float c = fg * creg[i] + ig * gg;
      creg[i] = c;
      float h = og * tanhf(c);
      unsigned short hb = f2bf(h);
      sm_h[p] = hb;                                   // p == b*256 + jx
      hout[((unsigned int)t << 14) + p] = hb;
    }
    __syncthreads();
  }
}

// logits = sigmoid(h_last @ fc_w^T + fc_b), one thread per batch row
__global__ __launch_bounds__(64) void fc_head(const unsigned short* __restrict__ h1,
                                              const float* __restrict__ fw,
                                              const float* __restrict__ fb,
                                              float* __restrict__ out) {
  int b = threadIdx.x;
  const unsigned short* hr = h1 + (S_LEN - 1) * (BATCH * HID) + b * HID;
  float s = 0.f;
  for (int j = 0; j < HID; ++j) s += bf2f(hr[j]) * fw[j];
  s += fb[0];
  out[b] = sigm(s);
}

// ---------------------------------------------------------------------------
extern "C" void kernel_launch(void* const* d_in, const int* in_sizes, int n_in,
                              void* d_out, int out_size, void* d_ws, size_t ws_size,
                              hipStream_t stream) {
  const float* x    = (const float*)d_in[0];   // [64,1,512,256]
  const float* W_ih = (const float*)d_in[1];   // [2,1024,256]
  const float* W_hh = (const float*)d_in[2];   // [2,1024,256]
  const float* b_ih = (const float*)d_in[3];   // [2,1024]
  const float* b_hh = (const float*)d_in[4];   // [2,1024]
  const float* fc_w = (const float*)d_in[5];   // [1,256]
  const float* fc_b = (const float*)d_in[6];   // [1]
  float* out = (float*)d_out;                  // [64,1]

  char* ws = (char*)d_ws;
  unsigned short* xb   = (unsigned short*)(ws);                        // 16 MB bf16 x, time-major
  unsigned short* h0   = (unsigned short*)(ws + (16ull << 20));        // 16 MB layer-0 h sequence
  unsigned short* h1   = (unsigned short*)(ws + (32ull << 20));        // 16 MB layer-1 h sequence
  unsigned int*   wf   = (unsigned int*)  (ws + (48ull << 20));        //  2 MB bf16 weight fragments
  float*          bias = (float*)         (ws + (50ull << 20));        //  8 KB fused biases

  pack_weights<<<2048, 256, 0, stream>>>(W_ih, W_hh, wf);
  pack_bias<<<8, 256, 0, stream>>>(b_ih, b_hh, bias);
  convert_x<<<32768, 256, 0, stream>>>(x, xb);

  const size_t lds_bytes = 164u * 1024u;  // 32KB h + 128KB gates + 4KB bias
  lstm_layer_k<<<1, 1024, lds_bytes, stream>>>(xb, wf,                         bias,        h0);
  lstm_layer_k<<<1, 1024, lds_bytes, stream>>>(h0, wf + 2u * FRAG_U32_PER_MAT, bias + 1024, h1);

  fc_head<<<1, 64, 0, stream>>>(h1, fc_w, fc_b, out);
}
